// SwitchSAE_20220706029913
// MI455X (gfx1250) — matrix operand order repaired and checked by reading, not compiled, not run on hardware
//
#include <hip/hip_runtime.h>
#include <hip/hip_bf16.h>

#define B_TOK 8192
#define D_IN  768
#define N_EXP 16
#define E_DIM 1024
#define MAX_MTILES 272   // sum of per-expert 32-aligned segments <= 272 tiles of 32 tokens

typedef __attribute__((ext_vector_type(16))) _Float16 v16h;
typedef __attribute__((ext_vector_type(8)))  _Float16 h8;
typedef __attribute__((ext_vector_type(4)))  _Float16 h4;
typedef __attribute__((ext_vector_type(8)))  float    v8f;

#define LDH 40   // LDS row stride in halfs (32 + 8 pad), 80 bytes, 16B-aligned rows

// ---- workspace layout (4-byte words) ----
#define WS_COUNTS   0                       // 16 int
#define WS_CURSORS  16                      // 16 int
#define WS_OFFSETS  32                      // 17 int
#define WS_TILEEXP  64                      // 512 int
#define WS_SUMP     576                     // 16 float
#define WS_IDXINT   640                     // 8192 int
#define WS_MAXPROB  (WS_IDXINT + B_TOK)     // 8192 float
#define WS_ORDER    (WS_MAXPROB + B_TOK)    // 8960 int (padded token order, -1 = hole)
#define WS_WASBITS  (WS_ORDER + 8960)       // 16*1024 int (float bits, values >= 0)

// -------------------------------------------------------------------------
__global__ void init_ws_kernel(int* __restrict__ ws) {
    int i = blockIdx.x * blockDim.x + threadIdx.x;
    if (i < 16) { ws[WS_COUNTS + i] = 0; ws[WS_CURSORS + i] = 0;
                  ((float*)ws)[WS_SUMP + i] = 0.0f; }
    if (i < 17)    ws[WS_OFFSETS + i] = 0;
    if (i < 512)   ws[WS_TILEEXP + i] = -1;
    if (i < 8960)  ws[WS_ORDER + i]   = -1;
    if (i < N_EXP * E_DIM) ws[WS_WASBITS + i] = 0;
}

// -------------------------------------------------------------------------
// One wave per token: lanes split K=768 for all 16 experts, xor-reduce,
// softmax; argmax prob == 1/sum(exp(l - max)).
__global__ void router_kernel(const float* __restrict__ act,
                              const float* __restrict__ router_b,
                              const float* __restrict__ router,
                              int* __restrict__ ws,
                              float* __restrict__ outIdx) {
    int wave = threadIdx.x >> 5;
    int lane = threadIdx.x & 31;
    int t = blockIdx.x * 8 + wave;
    if (t >= B_TOK) return;

    float acc[N_EXP];
#pragma unroll
    for (int e = 0; e < N_EXP; ++e) acc[e] = 0.0f;

    for (int k = lane; k < D_IN; k += 32) {
        float a = act[(size_t)t * D_IN + k] - router_b[k];
#pragma unroll
        for (int e = 0; e < N_EXP; ++e)
            acc[e] += a * router[k * N_EXP + e];
    }
#pragma unroll
    for (int e = 0; e < N_EXP; ++e) {
#pragma unroll
        for (int m = 16; m >= 1; m >>= 1)
            acc[e] += __shfl_xor(acc[e], m, 32);
    }

    float mx = acc[0]; int bi = 0;
#pragma unroll
    for (int e = 1; e < N_EXP; ++e)
        if (acc[e] > mx) { mx = acc[e]; bi = e; }

    float pr[N_EXP]; float se = 0.0f;
#pragma unroll
    for (int e = 0; e < N_EXP; ++e) { pr[e] = __expf(acc[e] - mx); se += pr[e]; }
    float inv = 1.0f / se;               // == max prob

    if (lane == 0) {
        float* sumP = (float*)ws + WS_SUMP;
#pragma unroll
        for (int e = 0; e < N_EXP; ++e) atomicAdd(&sumP[e], pr[e] * inv);
        atomicAdd(&ws[WS_COUNTS + bi], 1);
        ws[WS_IDXINT + t] = bi;
        ((float*)ws)[WS_MAXPROB + t] = inv;
        outIdx[t] = (float)bi;
    }
}

// -------------------------------------------------------------------------
__global__ void prep_kernel(int* __restrict__ ws) {
    if (threadIdx.x == 0 && blockIdx.x == 0) {
        int off = 0;
        ws[WS_OFFSETS + 0] = 0;
        for (int e = 0; e < N_EXP; ++e) {
            int padded = ((ws[WS_COUNTS + e] + 31) >> 5) << 5;
            off += padded;
            ws[WS_OFFSETS + e + 1] = off;
        }
        int ntiles = off >> 5;
        for (int tl = 0; tl < ntiles; ++tl) {
            int pos = tl << 5;
            int e = 0;
            while (e < N_EXP - 1 && pos >= ws[WS_OFFSETS + e + 1]) ++e;
            ws[WS_TILEEXP + tl] = e;
        }
    }
}

__global__ void scatter_kernel(int* __restrict__ ws) {
    int t = blockIdx.x * blockDim.x + threadIdx.x;
    if (t >= B_TOK) return;
    int e = ws[WS_IDXINT + t];
    int pos = atomicAdd(&ws[WS_CURSORS + e], 1);
    ws[WS_ORDER + ws[WS_OFFSETS + e] + pos] = t;
}

// -------------------------------------------------------------------------
// Fragment loads, all contiguous 16B LDS reads (-> ds_load_b128).
// A (16x32 f16): lane -> row M = lane&15; K runs [8h..8h+7] and [16+8h..23+8h]
// B (32x16 f16): lane -> col N = lane&15; K runs [8h..8h+7] and [16+8h..23+8h]
__device__ __forceinline__ v16h frag_a(const _Float16* __restrict__ s,
                                       int row, int half) {
    h8 lo = *(const h8*)(s + row * LDH + 8 * half);
    h8 hi = *(const h8*)(s + row * LDH + 16 + 8 * half);
    return __builtin_shufflevector(lo, hi, 0,1,2,3,4,5,6,7,8,9,10,11,12,13,14,15);
}
__device__ __forceinline__ v16h frag_b(const _Float16* __restrict__ s,
                                       int row, int half) {
    h8 lo = *(const h8*)(s + row * LDH + 8 * half);
    h8 hi = *(const h8*)(s + row * LDH + 16 + 8 * half);
    return __builtin_shufflevector(lo, hi, 0,1,2,3,4,5,6,7,8,9,10,11,12,13,14,15);
}
__device__ __forceinline__ h8 cvt8(float4 a, float4 b) {
    h8 r = { (_Float16)a.x, (_Float16)a.y, (_Float16)a.z, (_Float16)a.w,
             (_Float16)b.x, (_Float16)b.y, (_Float16)b.z, (_Float16)b.w };
    return r;
}

// -------------------------------------------------------------------------
// Encoder GEMM: D[n_latent(128), token(32)] = dec[e](rows) x x(tokens)^T
// latent = relu(D); scatter rows to full_latent; was_active via shfl-max.
__global__ void latent_kernel(const float* __restrict__ X,
                              const float* __restrict__ pre_b,
                              const float* __restrict__ decW,
                              int* __restrict__ ws,
                              float* __restrict__ latentOut) {
    __shared__ __align__(16) _Float16 sW[128 * LDH];  // weight rows x 32 K
    __shared__ __align__(16) _Float16 sX[32 * LDH];   // token rows x 32 K

    int e = ws[WS_TILEEXP + blockIdx.x];
    if (e < 0) return;
    int m0 = blockIdx.x << 5;   // token base (sorted order)
    int n0 = blockIdx.y << 7;   // latent slab base (128)
    const float* wE = decW + (size_t)e * E_DIM * D_IN;   // [1024][768] row=latent n

    int tid = threadIdx.x;
    int rW = tid >> 1, cW = (tid & 1) << 4;   // weights: 16 f32/thread
    int rX = tid >> 3, cX = (tid & 7) << 2;   // tokens:   4 f32/thread
    int tokLd = ws[WS_ORDER + m0 + rX];

    int lane = tid & 31, wave = tid >> 5;
    int mw = wave << 4;                       // wave's weight-row subtile
    int half = lane >> 4, ln = lane & 15;

    v8f acc0 = {}, acc1 = {};
    for (int k0 = 0; k0 < D_IN; k0 += 32) {
        __syncthreads();
        const float* wrow = wE + (size_t)(n0 + rW) * D_IN + k0 + cW;
        __builtin_prefetch(wrow + 32, 0, 3);
        const float4* wp = (const float4*)wrow;
        float4 w0 = wp[0], w1 = wp[1], w2 = wp[2], w3 = wp[3];
        *(h8*)&sW[rW * LDH + cW]     = cvt8(w0, w1);
        *(h8*)&sW[rW * LDH + cW + 8] = cvt8(w2, w3);

        float4 xv = make_float4(0.f, 0.f, 0.f, 0.f);
        if (tokLd >= 0)
            xv = *(const float4*)(X + (size_t)tokLd * D_IN + k0 + cX);
        h4 hx = { (_Float16)(xv.x - pre_b[k0 + cX + 0]),
                  (_Float16)(xv.y - pre_b[k0 + cX + 1]),
                  (_Float16)(xv.z - pre_b[k0 + cX + 2]),
                  (_Float16)(xv.w - pre_b[k0 + cX + 3]) };
        *(h4*)&sX[rX * LDH + cX] = hx;
        __syncthreads();

        v16h a  = frag_a(sW, mw + ln, half);
        v16h b0 = frag_b(sX, ln,      half);
        v16h b1 = frag_b(sX, 16 + ln, half);
        acc0 = __builtin_amdgcn_wmma_f32_16x16x32_f16(
            false, a, false, b0, (short)0, acc0, false, false);
        acc1 = __builtin_amdgcn_wmma_f32_16x16x32_f16(
            false, a, false, b1, (short)0, acc1, false, false);
    }

    int nbase = n0 + mw + 8 * half;     // 8 contiguous latent dims per lane
#pragma unroll
    for (int s = 0; s < 2; ++s) {
        v8f acc = s ? acc1 : acc0;
        int tok = ws[WS_ORDER + m0 + s * 16 + ln];
        float v[8];
#pragma unroll
        for (int r = 0; r < 8; ++r) { float t = acc[r]; v[r] = t > 0.f ? t : 0.f; }
        if (tok >= 0) {
            float* dst = latentOut + (size_t)tok * E_DIM + nbase;
            *(float4*)dst       = make_float4(v[0], v[1], v[2], v[3]);
            *(float4*)(dst + 4) = make_float4(v[4], v[5], v[6], v[7]);
        }
#pragma unroll
        for (int r = 0; r < 8; ++r) {   // max over the 16 token lanes
            float m = v[r];
#pragma unroll
            for (int msk = 8; msk >= 1; msk >>= 1)
                m = fmaxf(m, __shfl_xor(m, msk, 32));
            if (ln == 0)                // relu'd >= 0: int bits are monotone
                atomicMax(&ws[WS_WASBITS + e * E_DIM + nbase + r],
                          __float_as_int(m));
        }
    }
}

// -------------------------------------------------------------------------
// Decoder GEMM: D[d_in(128 slab), token(32)] = enc[e](rows, = dec^T) x latent^T
// recon = max_prob * D + pre_b.
__global__ void recon_kernel(const float* __restrict__ latentIn,
                             const float* __restrict__ encW,
                             const float* __restrict__ pre_b,
                             const int* __restrict__ ws,
                             float* __restrict__ reconOut) {
    __shared__ __align__(16) _Float16 sW[128 * LDH];
    __shared__ __align__(16) _Float16 sX[32 * LDH];

    int e = ws[WS_TILEEXP + blockIdx.x];
    if (e < 0) return;
    int m0 = blockIdx.x << 5;   // token base
    int n0 = blockIdx.y << 7;   // d_in slab base (128)
    const float* wE = encW + (size_t)e * D_IN * E_DIM;   // [768][1024] row=d

    int tid = threadIdx.x;
    int rW = tid >> 1, cW = (tid & 1) << 4;
    int rX = tid >> 3, cX = (tid & 7) << 2;
    int tokLd = ws[WS_ORDER + m0 + rX];

    int lane = tid & 31, wave = tid >> 5;
    int mw = wave << 4;
    int half = lane >> 4, ln = lane & 15;

    v8f acc0 = {}, acc1 = {};
    for (int k0 = 0; k0 < E_DIM; k0 += 32) {
        __syncthreads();
        const float* wrow = wE + (size_t)(n0 + rW) * E_DIM + k0 + cW;
        __builtin_prefetch(wrow + 32, 0, 3);
        const float4* wp = (const float4*)wrow;
        float4 w0 = wp[0], w1 = wp[1], w2 = wp[2], w3 = wp[3];
        *(h8*)&sW[rW * LDH + cW]     = cvt8(w0, w1);
        *(h8*)&sW[rW * LDH + cW + 8] = cvt8(w2, w3);

        float4 xv = make_float4(0.f, 0.f, 0.f, 0.f);
        if (tokLd >= 0)
            xv = *(const float4*)(latentIn + (size_t)tokLd * E_DIM + k0 + cX);
        h4 hx = { (_Float16)xv.x, (_Float16)xv.y, (_Float16)xv.z, (_Float16)xv.w };
        *(h4*)&sX[rX * LDH + cX] = hx;
        __syncthreads();

        v16h a  = frag_a(sW, mw + ln, half);
        v16h b0 = frag_b(sX, ln,      half);
        v16h b1 = frag_b(sX, 16 + ln, half);
        acc0 = __builtin_amdgcn_wmma_f32_16x16x32_f16(
            false, a, false, b0, (short)0, acc0, false, false);
        acc1 = __builtin_amdgcn_wmma_f32_16x16x32_f16(
            false, a, false, b1, (short)0, acc1, false, false);
    }

    int nbase = n0 + mw + 8 * half;     // 8 contiguous d_in dims per lane
    float4 pb0 = *(const float4*)(pre_b + nbase);
    float4 pb1 = *(const float4*)(pre_b + nbase + 4);
    const float* maxProb = (const float*)ws + WS_MAXPROB;
#pragma unroll
    for (int s = 0; s < 2; ++s) {
        v8f acc = s ? acc1 : acc0;
        int tok = ws[WS_ORDER + m0 + s * 16 + ln];
        if (tok >= 0) {
            float p = maxProb[tok];
            float* dst = reconOut + (size_t)tok * D_IN + nbase;
            *(float4*)dst = make_float4(p * acc[0] + pb0.x, p * acc[1] + pb0.y,
                                        p * acc[2] + pb0.z, p * acc[3] + pb0.w);
            *(float4*)(dst + 4) = make_float4(p * acc[4] + pb1.x, p * acc[5] + pb1.y,
                                              p * acc[6] + pb1.z, p * acc[7] + pb1.w);
        }
    }
}

// -------------------------------------------------------------------------
__global__ void finalize_kernel(const int* __restrict__ ws,
                                float* __restrict__ outWas,
                                float* __restrict__ outProp,
                                float* __restrict__ outWeight) {
    int i = blockIdx.x * blockDim.x + threadIdx.x;
    if (i < N_EXP * E_DIM) {
        float v = __int_as_float(ws[WS_WASBITS + i]);
        outWas[i] = (v > 0.001f) ? 1.0f : 0.0f;
    }
    if (i < N_EXP) {
        outProp[i]   = (float)ws[WS_COUNTS + i] / (float)B_TOK;
        outWeight[i] = ((const float*)ws)[WS_SUMP + i] / (float)B_TOK;
    }
}

// -------------------------------------------------------------------------
extern "C" void kernel_launch(void* const* d_in, const int* in_sizes, int n_in,
                              void* d_out, int out_size, void* d_ws, size_t ws_size,
                              hipStream_t stream) {
    const float* act      = (const float*)d_in[0];
    const float* pre_b    = (const float*)d_in[1];
    const float* enc      = (const float*)d_in[2];
    const float* dec      = (const float*)d_in[3];
    const float* router_b = (const float*)d_in[4];
    const float* router   = (const float*)d_in[5];

    float* out       = (float*)d_out;
    float* outRecon  = out;
    float* outLatent = outRecon + (size_t)B_TOK * D_IN;
    float* outWas    = outLatent + (size_t)B_TOK * E_DIM;
    float* outIdx    = outWas + N_EXP * E_DIM;
    float* outProp   = outIdx + B_TOK;
    float* outWeight = outProp + N_EXP;

    int* ws = (int*)d_ws;

    init_ws_kernel<<<(N_EXP * E_DIM + 255) / 256, 256, 0, stream>>>(ws);
    router_kernel<<<B_TOK / 8, 256, 0, stream>>>(act, router_b, router, ws, outIdx);
    prep_kernel<<<1, 64, 0, stream>>>(ws);
    scatter_kernel<<<B_TOK / 256, 256, 0, stream>>>(ws);
    latent_kernel<<<dim3(MAX_MTILES, E_DIM / 128), 256, 0, stream>>>(
        act, pre_b, dec, ws, outLatent);
    recon_kernel<<<dim3(MAX_MTILES, D_IN / 128), 256, 0, stream>>>(
        outLatent, enc, pre_b, ws, outRecon);
    finalize_kernel<<<(N_EXP * E_DIM + 255) / 256, 256, 0, stream>>>(
        ws, outWas, outProp, outWeight);
}
